// Block_1486058684721
// MI455X (gfx1250) — matrix operand order repaired and checked by reading, not compiled, not run on hardware
//
#include <hip/hip_runtime.h>
#include <hip/hip_bf16.h>

// ---------------------------------------------------------------------------
// Transformer block (pre-LN attention + MLP) for gfx1250 (MI455X).
// All GEMMs via v_wmma_f32_16x16x32_f16 (f16 in, f32 accumulate).
// Attention K-tiles staged with global_load_async_to_lds_b128 (ASYNCcnt).
// B=4, T=2048, E=1024, H=16, HS=64.
// ---------------------------------------------------------------------------

#define Bsz 4
#define Tsz 2048
#define Esz 1024
#define Hsz 16
#define HSz 64
#define Msz (Bsz * Tsz)   // 8192 rows

typedef __attribute__((ext_vector_type(16))) _Float16 v16h;
typedef __attribute__((ext_vector_type(8)))  _Float16 v8h;
typedef __attribute__((ext_vector_type(4)))  _Float16 v4h;
typedef __attribute__((ext_vector_type(8)))  float    v8f;

union F16x16 { v16h v; struct { v8h lo, hi; } p; };

// A-fragment (16x32 f16): lane m = lane%16 holds row m; halves i =
// K[half*16 + i]. Two 16-byte loads.
__device__ inline v16h load_a_frag(const _Float16* __restrict__ rowp, int half_) {
  F16x16 f;
  f.p.lo = *reinterpret_cast<const v8h*>(rowp + half_ * 16);
  f.p.hi = *reinterpret_cast<const v8h*>(rowp + half_ * 16 + 8);
  return f.v;
}

// B-fragment (32x16 f16): lane n = lane%16 holds column n; halves i =
// B[half*16+i][n].  With B transposed in LDS as BT[n][k], this is 32
// contiguous bytes at BT[n][half*16].
__device__ inline v16h load_b_frag(const _Float16* __restrict__ btrow, int half_) {
  F16x16 f;
  f.p.lo = *reinterpret_cast<const v8h*>(btrow + half_ * 16);
  f.p.hi = *reinterpret_cast<const v8h*>(btrow + half_ * 16 + 8);
  return f.v;
}

// Async copy of 32 contiguous bytes global -> LDS (per lane), ASYNCcnt-tracked.
// INST_OFFSET is added to BOTH the LDS and global addresses (ISA 15.18.3).
__device__ inline void async_copy_b32bytes(uint32_t lds_addr, const void* gptr) {
  asm volatile(
      "global_load_async_to_lds_b128 %0, %1, off\n\t"
      "global_load_async_to_lds_b128 %0, %1, off offset:16"
      :: "v"(lds_addr), "v"((uint64_t)(uintptr_t)gptr) : "memory");
}
__device__ inline void wait_async0() {
  asm volatile("s_wait_asynccnt 0x0" ::: "memory");
}

// ---------------------------------------------------------------------------
// Weight conversion kernels (run every call; deterministic)
// ---------------------------------------------------------------------------
__global__ void pack_qkv_kernel(const float* __restrict__ Wq,
                                const float* __restrict__ Wk,
                                const float* __restrict__ Wv,
                                _Float16* __restrict__ out) {
  // out is [E][3E] row-major (K x N).  Column n: n<1024 -> Q, <2048 -> K, else V.
  size_t i = (size_t)blockIdx.x * 256 + threadIdx.x;   // 1024*3072 total
  int n = (int)(i % 3072);
  int e = (int)(i / 3072);
  int m = n >> 10;            // which matrix
  int c = n & 1023;           // h*64 + d
  int h = c >> 6, d = c & 63;
  const float* W = (m == 0) ? Wq : (m == 1) ? Wk : Wv;
  out[i] = (_Float16)W[(size_t)h * Esz * HSz + (size_t)e * HSz + d];
}

__global__ void cast_f32_to_f16(const float* __restrict__ in,
                                _Float16* __restrict__ out) {
  size_t i = (size_t)blockIdx.x * 256 + threadIdx.x;
  out[i] = (_Float16)in[i];
}

// ---------------------------------------------------------------------------
// LayerNorm -> f16   (one 1024-wide row per 256-thread block)
// ---------------------------------------------------------------------------
__global__ __launch_bounds__(256) void ln_kernel(const float* __restrict__ x,
                                                 const float* __restrict__ g,
                                                 const float* __restrict__ b,
                                                 _Float16* __restrict__ out) {
  const int row = blockIdx.x;
  const int tid = threadIdx.x;
  const float4 v = reinterpret_cast<const float4*>(x + (size_t)row * Esz)[tid];
  float s  = v.x + v.y + v.z + v.w;
  float sq = v.x * v.x + v.y * v.y + v.z * v.z + v.w * v.w;
#pragma unroll
  for (int m = 16; m >= 1; m >>= 1) {
    s  += __shfl_xor(s,  m, 32);
    sq += __shfl_xor(sq, m, 32);
  }
  __shared__ float red[16];
  const int wid = tid >> 5, lane = tid & 31;
  if (lane == 0) { red[wid] = s; red[wid + 8] = sq; }
  __syncthreads();
  float S = 0.f, SQ = 0.f;
#pragma unroll
  for (int i = 0; i < 8; ++i) { S += red[i]; SQ += red[i + 8]; }
  const float mean = S * (1.0f / Esz);
  const float var  = SQ * (1.0f / Esz) - mean * mean;
  const float rs   = rsqrtf(var + 1e-5f);
  const float4 gg = reinterpret_cast<const float4*>(g)[tid];
  const float4 bb = reinterpret_cast<const float4*>(b)[tid];
  v4h o;
  o[0] = (_Float16)((v.x - mean) * rs * gg.x + bb.x);
  o[1] = (_Float16)((v.y - mean) * rs * gg.y + bb.y);
  o[2] = (_Float16)((v.z - mean) * rs * gg.z + bb.z);
  o[3] = (_Float16)((v.w - mean) * rs * gg.w + bb.w);
  reinterpret_cast<v4h*>(out + (size_t)row * Esz)[tid] = o;
}

// ---------------------------------------------------------------------------
// Generic WMMA GEMM:  out[M,N] = epilogue(A[M,K] * B[K,N])
//   8 waves as 4(M) x 2(N); wave tile 32x64; block tile 128x128; K step 32.
//   EPI 0: f16 store (no bias)
//   EPI 1: f32 store, + bias[col] + residual[row,col]
//   EPI 2: f16 store, relu(+ bias[col])
// ---------------------------------------------------------------------------
#define BT_LD 48   // padded LDS stride (halves): 96B = 6 x 16B
template <int EPI>
__global__ __launch_bounds__(256) void gemm_f16(
    const _Float16* __restrict__ A, int lda,
    const _Float16* __restrict__ B, int ldb,
    const float* __restrict__ bias,
    const float* __restrict__ res, int ldres,
    void* __restrict__ outp, int ldo,
    int K) {
  __shared__ alignas(16) _Float16 BT[128 * BT_LD];   // [n][k]
  const int tid   = threadIdx.x;
  const int wid   = tid >> 5;
  const int lane  = tid & 31;
  const int half_ = lane >> 4;
  const int ln    = lane & 15;
  const int wm    = wid & 3;          // M quadrant within block
  const int wn    = wid >> 2;         // N half within block
  const int row0  = blockIdx.y * 128 + wm * 32;
  const int col0  = blockIdx.x * 128 + wn * 64;

  v8f acc[2][4] = {{{}, {}, {}, {}}, {{}, {}, {}, {}}};
  const _Float16* Arow0 = A + (size_t)(row0 + ln) * lda;
  const _Float16* Arow1 = A + (size_t)(row0 + 16 + ln) * lda;

  for (int kc = 0; kc < K; kc += 32) {
    __syncthreads();
    {
      const int kk = tid >> 3;          // 0..31
      const int n0 = (tid & 7) * 16;    // 0..112
      const _Float16* bp = B + (size_t)(kc + kk) * ldb + blockIdx.x * 128 + n0;
      v8h b0 = *reinterpret_cast<const v8h*>(bp);
      v8h b1 = *reinterpret_cast<const v8h*>(bp + 8);
      if (kc + 32 < K) __builtin_prefetch(bp + (size_t)32 * ldb, 0, 0);
#pragma unroll
      for (int j = 0; j < 8; ++j) {
        BT[(n0 + j) * BT_LD + kk]     = b0[j];
        BT[(n0 + 8 + j) * BT_LD + kk] = b1[j];
      }
    }
    __syncthreads();
    v16h af0 = load_a_frag(Arow0 + kc, half_);
    v16h af1 = load_a_frag(Arow1 + kc, half_);
#pragma unroll
    for (int nt = 0; nt < 4; ++nt) {
      v16h bf = load_b_frag(&BT[(wn * 64 + nt * 16 + ln) * BT_LD], half_);
      acc[0][nt] = __builtin_amdgcn_wmma_f32_16x16x32_f16(
          false, af0, false, bf, (short)0, acc[0][nt], false, false);
      acc[1][nt] = __builtin_amdgcn_wmma_f32_16x16x32_f16(
          false, af1, false, bf, (short)0, acc[1][nt], false, false);
    }
  }

#pragma unroll
  for (int mt = 0; mt < 2; ++mt) {
#pragma unroll
    for (int nt = 0; nt < 4; ++nt) {
#pragma unroll
      for (int r = 0; r < 8; ++r) {
        const int row = row0 + mt * 16 + half_ * 8 + r;  // C/D: row = half*8+r
        const int col = col0 + nt * 16 + ln;             //      col = lane%16
        float v = acc[mt][nt][r];
        if (EPI == 0) {
          reinterpret_cast<_Float16*>(outp)[(size_t)row * ldo + col] = (_Float16)v;
        } else if (EPI == 1) {
          v += bias[col] + res[(size_t)row * ldres + col];
          reinterpret_cast<float*>(outp)[(size_t)row * ldo + col] = v;
        } else {
          v += bias[col];
          v = v > 0.0f ? v : 0.0f;
          reinterpret_cast<_Float16*>(outp)[(size_t)row * ldo + col] = (_Float16)v;
        }
      }
    }
  }
}

// ---------------------------------------------------------------------------
// Flash attention: one block = 128 query rows of one (b,h).
// qkv is [8192][3072] f16: cols 0..1023 Q, 1024..2047 K, 2048..3071 V
// (head-major: col = h*64 + d).  Output o16 [8192][1024] f16.
// ---------------------------------------------------------------------------
#define VT_LD 72   // padded stride for VT (halves): 144B, odd multiple of 16B
__global__ __launch_bounds__(256) void attn_kernel(
    const _Float16* __restrict__ qkv, _Float16* __restrict__ o16) {
  __shared__ alignas(16) _Float16 Klds[64 * 64];       // [s][d] == BT for Q*K^T
  __shared__ alignas(16) _Float16 VT[64 * VT_LD];      // [d][s] == BT for P*V
  __shared__ alignas(16) _Float16 Plds[8 * 16 * 64];   // per-wave P patch

  const int tid = threadIdx.x, wid = tid >> 5, lane = tid & 31;
  const int half_ = lane >> 4, ln = lane & 15;
  const int bh = blockIdx.y;
  const int b = bh >> 4, h = bh & 15;
  const int t0 = blockIdx.x * 128;

  const size_t ldq = 3072;
  const _Float16* qbase = qkv + (size_t)b * Tsz * ldq + h * 64;
  const _Float16* kbase = qbase + 1024;
  const _Float16* vbase = qbase + 2048;

  // Q fragments for this wave's 16 rows (registers for all key blocks)
  const _Float16* qp = qbase + (size_t)(t0 + wid * 16 + ln) * ldq;
  const v16h qf0 = load_a_frag(qp, half_);
  const v16h qf1 = load_a_frag(qp + 32, half_);

  v8f oacc[4] = {{}, {}, {}, {}};
  float runmax[8], runsum[8];
#pragma unroll
  for (int r = 0; r < 8; ++r) { runmax[r] = -1e30f; runsum[r] = 0.0f; }
  const float scale = 0.125f;   // HS^-0.5

  const int s_  = tid >> 2;          // 0..63  (staging role)
  const int d0_ = (tid & 3) * 16;    // 0,16,32,48
  const uint32_t klds_addr =
      (uint32_t)(uintptr_t)&Klds[s_ * 64 + d0_];

  for (int sb = 0; sb < Tsz; sb += 64) {
    __syncthreads();
    // K tile: straight gather-copy -> async global->LDS (ASYNCcnt path)
    async_copy_b32bytes(klds_addr, kbase + (size_t)(sb + s_) * ldq + d0_);
    // V tile: transpose into VT (padded stride) with normal loads/stores
    {
      const _Float16* vp = vbase + (size_t)(sb + s_) * ldq + d0_;
      v8h v0 = *reinterpret_cast<const v8h*>(vp);
      v8h v1 = *reinterpret_cast<const v8h*>(vp + 8);
#pragma unroll
      for (int j = 0; j < 8; ++j) {
        VT[(d0_ + j) * VT_LD + s_]     = v0[j];
        VT[(d0_ + 8 + j) * VT_LD + s_] = v1[j];
      }
    }
    wait_async0();
    __syncthreads();

    // S = Q * K^T  (16x64 per wave, K-dim = 64 in two 32-chunks)
    v8f sacc[4] = {{}, {}, {}, {}};
#pragma unroll
    for (int st = 0; st < 4; ++st) {
      const _Float16* bt = &Klds[(st * 16 + ln) * 64];
      v16h bf0 = load_b_frag(bt, half_);
      v16h bf1 = load_b_frag(bt + 32, half_);
      sacc[st] = __builtin_amdgcn_wmma_f32_16x16x32_f16(
          false, qf0, false, bf0, (short)0, sacc[st], false, false);
      sacc[st] = __builtin_amdgcn_wmma_f32_16x16x32_f16(
          false, qf1, false, bf1, (short)0, sacc[st], false, false);
    }

    // Online softmax.  In C/D layout, row (half*8+r) spans 16 lanes of a
    // half, so xor-shuffles with masks 8..1 reduce over that row's columns.
    _Float16* P = &Plds[wid * 16 * 64];
#pragma unroll
    for (int r = 0; r < 8; ++r) {
      float m = -1e30f;
#pragma unroll
      for (int st = 0; st < 4; ++st) m = fmaxf(m, sacc[st][r] * scale);
#pragma unroll
      for (int msk = 8; msk >= 1; msk >>= 1) m = fmaxf(m, __shfl_xor(m, msk, 32));
      const float nm   = fmaxf(runmax[r], m);
      const float corr = __expf(runmax[r] - nm);
      float rs = 0.0f;
#pragma unroll
      for (int st = 0; st < 4; ++st) {
        const float p = __expf(sacc[st][r] * scale - nm);
        rs += p;
        P[(half_ * 8 + r) * 64 + st * 16 + ln] = (_Float16)p;
      }
#pragma unroll
      for (int msk = 8; msk >= 1; msk >>= 1) rs += __shfl_xor(rs, msk, 32);
      runsum[r] = runsum[r] * corr + rs;
      runmax[r] = nm;
#pragma unroll
      for (int dt = 0; dt < 4; ++dt) oacc[dt][r] *= corr;
    }

    // O += P * V  (P reloaded from LDS in A-fragment layout)
#pragma unroll
    for (int kc2 = 0; kc2 < 2; ++kc2) {
      v16h pf = load_a_frag(&P[ln * 64 + kc2 * 32], half_);
#pragma unroll
      for (int dt = 0; dt < 4; ++dt) {
        v16h bf = load_b_frag(&VT[(dt * 16 + ln) * VT_LD + kc2 * 32], half_);
        oacc[dt] = __builtin_amdgcn_wmma_f32_16x16x32_f16(
            false, pf, false, bf, (short)0, oacc[dt], false, false);
      }
    }
  }

  // Normalize and store (heads concatenated: col = h*64 + d)
#pragma unroll
  for (int r = 0; r < 8; ++r) {
    const float inv = 1.0f / runsum[r];
    const int trow = t0 + wid * 16 + half_ * 8 + r;
    _Float16* op = o16 + (size_t)(b * Tsz + trow) * Esz + h * 64;
#pragma unroll
    for (int dt = 0; dt < 4; ++dt)
      op[dt * 16 + ln] = (_Float16)(oacc[dt][r] * inv);
  }
}

// ---------------------------------------------------------------------------
// Launch
// ---------------------------------------------------------------------------
extern "C" void kernel_launch(void* const* d_in, const int* in_sizes, int n_in,
                              void* d_out, int out_size, void* d_ws, size_t ws_size,
                              hipStream_t stream) {
  const float* x     = (const float*)d_in[0];
  const float* ln1_g = (const float*)d_in[1];
  const float* ln1_b = (const float*)d_in[2];
  const float* Wq    = (const float*)d_in[3];
  const float* Wk    = (const float*)d_in[4];
  const float* Wv    = (const float*)d_in[5];
  const float* Wo    = (const float*)d_in[6];
  const float* bo    = (const float*)d_in[7];
  const float* ln2_g = (const float*)d_in[8];
  const float* ln2_b = (const float*)d_in[9];
  const float* W1    = (const float*)d_in[10];
  const float* b1    = (const float*)d_in[11];
  const float* W2    = (const float*)d_in[12];
  const float* b2    = (const float*)d_in[13];

  char* ws = (char*)d_ws;
  size_t off = 0;
  auto alloc = [&](size_t bytes) {
    void* p = ws + off;
    off += (bytes + 255) & ~(size_t)255;
    return p;
  };
  _Float16* wQKV = (_Float16*)alloc((size_t)Esz * 3 * Esz * 2);       // [1024][3072]
  _Float16* wO   = (_Float16*)alloc((size_t)Esz * Esz * 2);           // [1024][1024]
  _Float16* wW1  = (_Float16*)alloc((size_t)Esz * 4 * Esz * 2);       // [1024][4096]
  _Float16* wW2  = (_Float16*)alloc((size_t)4 * Esz * Esz * 2);       // [4096][1024]
  _Float16* h16  = (_Float16*)alloc((size_t)Msz * Esz * 2);           // LN1 out
  _Float16* qkv  = (_Float16*)alloc((size_t)Msz * 3 * Esz * 2);       // QKV
  _Float16* o16  = (_Float16*)alloc((size_t)Msz * Esz * 2);           // attn out
  float*    x1   = (float*)   alloc((size_t)Msz * Esz * 4);           // post-attn residual
  _Float16* h2   = (_Float16*)alloc((size_t)Msz * Esz * 2);           // LN2 out
  _Float16* hid  = (_Float16*)alloc((size_t)Msz * 4 * Esz * 2);       // MLP hidden
  (void)ws_size; (void)in_sizes; (void)n_in; (void)out_size;

  // Weight conversion
  pack_qkv_kernel<<<(Esz * 3 * Esz) / 256, 256, 0, stream>>>(Wq, Wk, Wv, wQKV);
  cast_f32_to_f16<<<(Esz * Esz) / 256, 256, 0, stream>>>(Wo, wO);
  cast_f32_to_f16<<<(Esz * 4 * Esz) / 256, 256, 0, stream>>>(W1, wW1);
  cast_f32_to_f16<<<(4 * Esz * Esz) / 256, 256, 0, stream>>>(W2, wW2);

  // h = LN1(x)
  ln_kernel<<<Msz, 256, 0, stream>>>(x, ln1_g, ln1_b, h16);

  // qkv = h @ [Wq|Wk|Wv]   (8192 x 1024 x 3072)
  gemm_f16<0><<<dim3(3 * Esz / 128, Msz / 128), 256, 0, stream>>>(
      h16, Esz, wQKV, 3 * Esz, nullptr, nullptr, 0, qkv, 3 * Esz, Esz);

  // attention
  attn_kernel<<<dim3(Tsz / 128, Bsz * Hsz), 256, 0, stream>>>(qkv, o16);

  // x1 = x + o @ Wo + bo   (8192 x 1024 x 1024)
  gemm_f16<1><<<dim3(Esz / 128, Msz / 128), 256, 0, stream>>>(
      o16, Esz, wO, Esz, bo, x, Esz, x1, Esz, Esz);

  // h2 = LN2(x1)
  ln_kernel<<<Msz, 256, 0, stream>>>(x1, ln2_g, ln2_b, h2);

  // hid = relu(h2 @ W1 + b1)   (8192 x 1024 x 4096)
  gemm_f16<2><<<dim3(4 * Esz / 128, Msz / 128), 256, 0, stream>>>(
      h2, Esz, wW1, 4 * Esz, b1, nullptr, 0, hid, 4 * Esz, Esz);

  // out = x1 + hid @ W2 + b2   (8192 x 4096 x 1024)
  gemm_f16<1><<<dim3(Esz / 128, Msz / 128), 256, 0, stream>>>(
      hid, 4 * Esz, wW2, Esz, b2, x1, Esz, d_out, Esz, 4 * Esz);
}
